// Head_24661702213738
// MI455X (gfx1250) — compile-verified, matrix operand order checked by reading, not run on hardware
//
#include <hip/hip_runtime.h>
#include <hip/hip_bf16.h>
#include <math.h>

// ---------------------------------------------------------------------------
// Attention head: out = softmax((x Wq^T)(x Wk^T)^T / sqrt(D)) @ x @ Wv^T
// N=8192, D=1024, fp32 in/out. All GEMMs in bf16 WMMA (v_wmma_f32_16x16x32_bf16),
// flash-style online softmax so the 8192x8192 logits are never materialized.
// ---------------------------------------------------------------------------

#define NQ 8192
#define DD 1024

typedef __attribute__((ext_vector_type(16))) __bf16 v16bf;
typedef __attribute__((ext_vector_type(8)))  float  v8f;

union frag_u { uint4 u[2]; v16bf v; };
union h16_u  { __bf16 h[16]; uint4 u[2]; };

// Row-contiguous fragment load (works for A tiles and for B tiles that are
// "multiply by transposed row-major matrix": lane -> row, 2x16B chunks along K).
// Per ISA 16-bit A layout: lanes 0-15 hold K {0..7,16..23}, lanes 16-31 hold
// K {8..15,24..31} of the 32-deep K tile.
__device__ __forceinline__ v16bf load_frag_row(const __bf16* base, int ld,
                                               int r0, int k0, int lane) {
    int half = lane >> 4;
    int lr   = lane & 15;
    const __bf16* p = base + (size_t)(r0 + lr) * (size_t)ld + k0 + half * 8;
    frag_u f;
    f.u[0] = *(const uint4*)(p);        // K = k0+half*8 .. +7
    f.u[1] = *(const uint4*)(p + 16);   // K = k0+16+half*8 .. +7
    return f.v;
}

__device__ __forceinline__ v8f wmma_bf16(v16bf a, v16bf b, v8f c) {
    return __builtin_amdgcn_wmma_f32_16x16x32_bf16(
        false, a, false, b, (short)0, c, false, false);
}

// ---------------------------------------------------------------------------
// Stage 0a: fp32 -> bf16 conversion (weights)
// ---------------------------------------------------------------------------
__global__ void cvt_kernel(const float* __restrict__ src,
                           __bf16* __restrict__ dst, int n) {
    int i = blockIdx.x * blockDim.x + threadIdx.x;
    if (i < n) dst[i] = (__bf16)src[i];
}

// ---------------------------------------------------------------------------
// Stage 0b: x -> Xh (row major bf16) and XhT (transposed [D][N] bf16).
// LDS-tiled 64x64 transpose: coalesced float4 reads, packed 32B writes.
// Grid: (D/64, N/64), 256 threads.
// ---------------------------------------------------------------------------
__global__ __launch_bounds__(256)
void cvt_x_kernel(const float* __restrict__ x,
                  __bf16* __restrict__ Xh,
                  __bf16* __restrict__ XhT) {
    __shared__ __bf16 tile[64][72];   // padded to dodge bank conflicts

    int t  = threadIdx.x;
    int n0 = blockIdx.y * 64;
    int d0 = blockIdx.x * 64;

    // read phase: thread -> (row r, 16-col chunk cq) of the 64x64 tile
    int r  = t >> 2;
    int cq = (t & 3) * 16;
    const float* src = x + (size_t)(n0 + r) * DD + d0 + cq;
    h16_u tmp;
#pragma unroll
    for (int i = 0; i < 16; i += 4) {
        float4 f = *(const float4*)(src + i);
        tmp.h[i + 0] = (__bf16)f.x; tmp.h[i + 1] = (__bf16)f.y;
        tmp.h[i + 2] = (__bf16)f.z; tmp.h[i + 3] = (__bf16)f.w;
    }
    __bf16* xr = Xh + (size_t)(n0 + r) * DD + d0 + cq;
    *(uint4*)(xr)     = tmp.u[0];
    *(uint4*)(xr + 8) = tmp.u[1];
#pragma unroll
    for (int i = 0; i < 16; ++i) tile[r][cq + i] = tmp.h[i];
    __syncthreads();

    // write phase: thread -> (col dloc, 16-row chunk nq); packed 32B rows of XhT
    int dloc = t >> 2;
    int nq   = (t & 3) * 16;
    h16_u out;
#pragma unroll
    for (int i = 0; i < 16; ++i) out.h[i] = tile[nq + i][dloc];
    __bf16* dst = XhT + (size_t)(d0 + dloc) * NQ + n0 + nq;
    *(uint4*)(dst)     = out.u[0];
    *(uint4*)(dst + 8) = out.u[1];
}

// ---------------------------------------------------------------------------
// Stage 1/3: C[M][Nc] = scale * A[M][K] @ B[Nc][K]^T   (bf16 in, bf16/f32 out)
// Block: 128 threads (4 waves). Block tile 64x128, wave tile 16x128
// (A fragment reused across 8 WMMAs per k-step).
// ---------------------------------------------------------------------------
template <bool BF16OUT>
__global__ __launch_bounds__(128)
void gemm_bt_kernel(const __bf16* __restrict__ A, const __bf16* __restrict__ B,
                    void* __restrict__ C, int M, int Nc, int K, float scale) {
    int tid  = threadIdx.x;
    int wave = tid >> 5;
    int lane = tid & 31;
    int half = lane >> 4;
    int lr   = lane & 15;

    int m0 = blockIdx.y * 64 + wave * 16;
    int n0 = blockIdx.x * 128;

    v8f c[8];
    for (int f = 0; f < 8; ++f)
        for (int r = 0; r < 8; ++r) c[f][r] = 0.0f;

    for (int k0 = 0; k0 < K; k0 += 32) {
        v16bf a = load_frag_row(A, K, m0, k0, lane);
#pragma unroll
        for (int f = 0; f < 8; ++f) {
            v16bf b = load_frag_row(B, K, n0 + f * 16, k0, lane);
            c[f] = wmma_bf16(a, b, c[f]);
        }
    }

#pragma unroll
    for (int f = 0; f < 8; ++f) {
#pragma unroll
        for (int r = 0; r < 8; ++r) {
            int row = m0 + r + half * 8;
            int col = n0 + f * 16 + lr;
            float v = scale * c[f][r];
            if (BF16OUT)
                ((__bf16*)C)[(size_t)row * Nc + col] = (__bf16)v;
            else
                ((float*)C)[(size_t)row * Nc + col] = v;
        }
    }
}

// ---------------------------------------------------------------------------
// Stage 2: flash attention core. Oh[m][d] = softmax_row(Q K^T) @ X  (bf16 out)
// Grid: N/16 blocks, 256 threads (8 waves). Block owns 16 query rows.
// Wave w owns D-slice [w*128, w*128+128) for both partial-S and O accumulation.
// KV streamed in tiles of 64. Softmax is fully parallel: thread = (row,
// 4-column group); per-row reductions via __shfl_xor within a 16-lane half
// (rows never straddle wave32 halves given the C-fragment layout).
// ---------------------------------------------------------------------------
#define BN 64

__global__ __launch_bounds__(256)
void flash_kernel(const __bf16* __restrict__ Qh, const __bf16* __restrict__ Kh,
                  const __bf16* __restrict__ XhT, __bf16* __restrict__ Oh) {
    __shared__ float  s_part[8][16][BN];       // 32 KB partial logits
    __shared__ __bf16 p_tile[16][BN + 8];      // softmaxed P tile (padded)
    __shared__ float  m_run[16], l_run[16], cfac[16];

    int tid  = threadIdx.x;
    int wave = tid >> 5;
    int lane = tid & 31;
    int half = lane >> 4;
    int lr   = lane & 15;

    int m0 = blockIdx.x * 16;      // query rows
    int d0 = wave * 128;           // this wave's D slice

    // Q fragments for this wave's D slice: 4 x (16x32)  (scale folded into Qh)
    v16bf qf[4];
#pragma unroll
    for (int i = 0; i < 4; ++i)
        qf[i] = load_frag_row(Qh, DD, m0, d0 + 32 * i, lane);

    // O accumulator: 16 rows x 128 cols fp32 -> 8 C fragments
    v8f of[8];
    for (int j = 0; j < 8; ++j)
        for (int r = 0; r < 8; ++r) of[j][r] = 0.0f;

    if (tid < 16) { m_run[tid] = -INFINITY; l_run[tid] = 0.0f; }
    __syncthreads();

    for (int kv0 = 0; kv0 < NQ; kv0 += BN) {
        // prefetch next KV tile (global_prefetch_b8)
        if (kv0 + BN < NQ) {
            __builtin_prefetch(Kh + (size_t)(kv0 + BN + lr) * DD + d0 + half * 8,
                               0, 1);
            __builtin_prefetch(XhT + (size_t)(d0 + lr) * NQ + kv0 + BN + half * 8,
                               0, 1);
        }

        // ---- partial S over this wave's D slice: S[16][64] ----
        v8f sf[4];
        for (int f = 0; f < 4; ++f)
            for (int r = 0; r < 8; ++r) sf[f][r] = 0.0f;
#pragma unroll
        for (int i = 0; i < 4; ++i) {
            v16bf a = qf[i];
#pragma unroll
            for (int f = 0; f < 4; ++f) {
                v16bf b = load_frag_row(Kh, DD, kv0 + f * 16, d0 + 32 * i, lane);
                sf[f] = wmma_bf16(a, b, sf[f]);
            }
        }
#pragma unroll
        for (int f = 0; f < 4; ++f)
#pragma unroll
            for (int r = 0; r < 8; ++r)
                s_part[wave][r + half * 8][f * 16 + lr] = sf[f][r];
        __syncthreads();

        // ---- fused 8-way reduce + online softmax (all 256 threads) ----
        {
            int row   = tid >> 4;         // 0..15; rows 2w,2w+1 owned by wave w
            int cg    = tid & 15;         // column group within the row
            int cbase = cg * 4;
            float sv[4];
#pragma unroll
            for (int c = 0; c < 4; ++c) {
                float acc = 0.0f;
#pragma unroll
                for (int w = 0; w < 8; ++w) acc += s_part[w][row][cbase + c];
                sv[c] = acc;
            }
            float tmax = fmaxf(fmaxf(sv[0], sv[1]), fmaxf(sv[2], sv[3]));
#pragma unroll
            for (int m = 1; m < 16; m <<= 1)
                tmax = fmaxf(tmax, __shfl_xor(tmax, m, 32));
            float m_old = m_run[row];
            float m_new = fmaxf(m_old, tmax);
            float corr  = __expf(m_old - m_new);
            float psum  = 0.0f;
#pragma unroll
            for (int c = 0; c < 4; ++c) {
                float p = __expf(sv[c] - m_new);
                psum += p;
                p_tile[row][cbase + c] = (__bf16)p;
            }
#pragma unroll
            for (int m = 1; m < 16; m <<= 1)
                psum += __shfl_xor(psum, m, 32);
            if (cg == 0) {
                m_run[row] = m_new;
                l_run[row] = corr * l_run[row] + psum;
                cfac[row]  = corr;
            }
        }
        __syncthreads();

        // ---- rescale O by correction factor, then O += P @ X_kv ----
        float cf8[8];
#pragma unroll
        for (int r = 0; r < 8; ++r) cf8[r] = cfac[r + half * 8];
#pragma unroll
        for (int j = 0; j < 8; ++j)
#pragma unroll
            for (int r = 0; r < 8; ++r) of[j][r] *= cf8[r];

        // P A-fragments from LDS (2 x 16x32)
        v16bf pa[2];
#pragma unroll
        for (int i2 = 0; i2 < 2; ++i2) {
            const __bf16* pp = &p_tile[lr][32 * i2 + half * 8];
            frag_u f;
            f.u[0] = *(const uint4*)(pp);
            f.u[1] = *(const uint4*)(pp + 16);
            pa[i2] = f.v;
        }
        // B[k][n] = X[kv0+k][d0+j*16+n] = XhT[d0+j*16+n][kv0+k]  (row-contig!)
#pragma unroll
        for (int j = 0; j < 8; ++j) {
#pragma unroll
            for (int i2 = 0; i2 < 2; ++i2) {
                v16bf b = load_frag_row(XhT, NQ, d0 + j * 16, kv0 + 32 * i2, lane);
                of[j] = wmma_bf16(pa[i2], b, of[j]);
            }
        }
        __syncthreads();
    }

    // ---- finalize: divide by row sums, store bf16 ----
    float lf8[8];
#pragma unroll
    for (int r = 0; r < 8; ++r) lf8[r] = 1.0f / l_run[r + half * 8];
#pragma unroll
    for (int j = 0; j < 8; ++j)
#pragma unroll
        for (int r = 0; r < 8; ++r) {
            int row = m0 + r + half * 8;
            int col = d0 + j * 16 + lr;
            Oh[(size_t)row * DD + col] = (__bf16)(of[j][r] * lf8[r]);
        }
}

// ---------------------------------------------------------------------------
// Host-side launcher
// ---------------------------------------------------------------------------
extern "C" void kernel_launch(void* const* d_in, const int* in_sizes, int n_in,
                              void* d_out, int out_size, void* d_ws, size_t ws_size,
                              hipStream_t stream) {
    const float* x  = (const float*)d_in[0];
    const float* Wq = (const float*)d_in[1];
    const float* Wk = (const float*)d_in[2];
    const float* Wv = (const float*)d_in[3];

    char* ws = (char*)d_ws;
    const size_t ND  = (size_t)NQ * DD;   // 8M elems
    const size_t DDs = (size_t)DD * DD;   // 1M elems

    __bf16* Xh  = (__bf16*)(ws);
    __bf16* XhT = (__bf16*)(ws + 2 * ND);
    __bf16* Qh  = (__bf16*)(ws + 4 * ND);
    __bf16* Kh  = (__bf16*)(ws + 6 * ND);
    __bf16* Oh  = (__bf16*)(ws + 8 * ND);
    __bf16* Wqh = (__bf16*)(ws + 10 * ND);
    __bf16* Wkh = (__bf16*)(ws + 10 * ND + 2 * DDs);
    __bf16* Wvh = (__bf16*)(ws + 10 * ND + 4 * DDs);

    // Stage 0: conversions (tiled transpose for XhT)
    dim3 gx(DD / 64, NQ / 64);
    cvt_x_kernel<<<gx, 256, 0, stream>>>(x, Xh, XhT);
    cvt_kernel<<<(int)(DDs / 256), 256, 0, stream>>>(Wq, Wqh, (int)DDs);
    cvt_kernel<<<(int)(DDs / 256), 256, 0, stream>>>(Wk, Wkh, (int)DDs);
    cvt_kernel<<<(int)(DDs / 256), 256, 0, stream>>>(Wv, Wvh, (int)DDs);

    // Stage 1: projections (1/sqrt(D) folded into Q)
    dim3 gq(DD / 128, NQ / 64);
    gemm_bt_kernel<true><<<gq, 128, 0, stream>>>(Xh, Wqh, Qh, NQ, DD, DD,
                                                 1.0f / sqrtf((float)DD));
    gemm_bt_kernel<true><<<gq, 128, 0, stream>>>(Xh, Wkh, Kh, NQ, DD, DD, 1.0f);

    // Stage 2: flash attention -> Oh = softmax(Q K^T) @ X
    flash_kernel<<<NQ / 16, 256, 0, stream>>>(Qh, Kh, XhT, Oh);

    // Stage 3: output projection, fp32 out
    gemm_bt_kernel<false><<<gq, 128, 0, stream>>>(Oh, Wvh, d_out, NQ, DD, DD, 1.0f);

    (void)in_sizes; (void)n_in; (void)out_size; (void)ws_size;
}
